// MyConv2D_44822278701233
// MI455X (gfx1250) — matrix operand order repaired and checked by reading, not compiled, not run on hardware
//
#include <hip/hip_runtime.h>

// ---- CDNA5 vector types ----
typedef __attribute__((ext_vector_type(16))) _Float16 v16h;
typedef __attribute__((ext_vector_type(8)))  _Float16 v8h;
typedef __attribute__((ext_vector_type(8)))  float    v8f;

#define KH    3
#define KW    3
#define CIN   128
#define FOUT  256
#define HO    62
#define WO    62
#define KDIM  (KH * KW * CIN)     // 1152
#define KSTEPS (KDIM / 32)        // 36 k-steps of depth 32
#define LDSS  40                  // LDS row stride in f16 (80B, 16B aligned, bank-skewed)
#define NIN   (32 * 64 * 64 * CIN)   // 16,777,216 input elements
#define NW    (FOUT * KDIM)          // 294,912 weight elements

// Split 16 fp32 values into hi/lo f16 halves and store both to LDS as 2x b128 each.
__device__ __forceinline__ void split_store16(const float* __restrict__ v,
                                              _Float16* __restrict__ hiD,
                                              _Float16* __restrict__ loD) {
    v8h h0, h1, l0, l1;
#pragma unroll
    for (int k = 0; k < 8; ++k) {
        _Float16 ha = (_Float16)v[k];
        _Float16 hb = (_Float16)v[k + 8];
        h0[k] = ha;
        h1[k] = hb;
        l0[k] = (_Float16)(v[k]     - (float)ha);
        l1[k] = (_Float16)(v[k + 8] - (float)hb);
    }
    *(v8h*)(hiD)     = h0;
    *(v8h*)(hiD + 8) = h1;
    *(v8h*)(loD)     = l0;
    *(v8h*)(loD + 8) = l1;
}

__device__ __forceinline__ v16h cat8(v8h a, v8h b) {
    return __builtin_shufflevector(a, b, 0, 1, 2, 3, 4, 5, 6, 7,
                                          8, 9, 10, 11, 12, 13, 14, 15);
}

// ---- Pre-pass: x -> (f16 hi, f16 lo) exactly once per element ----
__global__ __launch_bounds__(256)
void split_f32_to_f16x2(const float* __restrict__ src,
                        _Float16* __restrict__ hi,
                        _Float16* __restrict__ lo, int n8) {
    const int gid = blockIdx.x * 256 + threadIdx.x;
    if (gid >= n8) return;
    const long base = (long)gid * 8;
    float4 v0 = ((const float4*)(src + base))[0];
    float4 v1 = ((const float4*)(src + base))[1];
    float f[8] = { v0.x, v0.y, v0.z, v0.w, v1.x, v1.y, v1.z, v1.w };
    v8h h, l;
#pragma unroll
    for (int k = 0; k < 8; ++k) {
        _Float16 hh = (_Float16)f[k];
        h[k] = hh;
        l[k] = (_Float16)(f[k] - (float)hh);
    }
    *(v8h*)(hi + base) = h;
    *(v8h*)(lo + base) = l;
}

// ---- Main implicit-GEMM conv: 128(M) x 256(F) tile per workgroup ----
// MODE 0: fp32 global loads, hi/lo split at LDS-store time (fallback).
// MODE 1: operands pre-split into d_ws; hot loop is pure f16 copies + WMMA.
template <int MODE>
__global__ __launch_bounds__(256)
void conv3x3_wmma_f16split_kernel(const float* __restrict__ in,
                                  const float* __restrict__ w,
                                  const float* __restrict__ bias,
                                  float* __restrict__ out,
                                  const _Float16* __restrict__ inHi,
                                  const _Float16* __restrict__ inLo,
                                  const _Float16* __restrict__ wHi,
                                  const _Float16* __restrict__ wLo) {
    // LDS: A 128x32 + B 256x32, hi/lo f16, stride-40 rows. 60 KB total.
    __shared__ _Float16 AsHi[128 * LDSS];
    __shared__ _Float16 AsLo[128 * LDSS];
    __shared__ _Float16 BsHi[256 * LDSS];
    __shared__ _Float16 BsLo[256 * LDSS];

    const int t     = threadIdx.x;
    const int lane  = t & 31;
    const int wave  = t >> 5;       // 0..7
    const int wm    = wave & 1;     // 2 waves along M (64 rows each)
    const int wn    = wave >> 1;    // 4 waves along F (64 cols each)
    const int mlane = lane & 15;
    const int g     = lane >> 4;    // half-wave group

    const int m0 = blockIdx.x * 128;   // 961 blocks; 961*128 == 123008 exactly

    // A loader: each thread owns 16 k-elements of one patch row.
    const int aRow = t >> 1;          // 0..127
    const int cOff = (t & 1) * 16;    // 0 or 16
    const int r  = m0 + aRow;
    const int n  = r / (HO * WO);
    const int rm = r - n * (HO * WO);
    const int ho = rm / WO;
    const int wo = rm - ho * WO;
    const long aBaseIdx = (long)((n * 64 + ho) * 64 + wo) * CIN;
    // B loader: each thread owns all 32 k-elements of one filter row (t = f).
    const long bBaseIdx = (long)t * KDIM;

    v8f acc[4][4] = {};   // 4 M-subtiles x 4 F-subtiles of 16x16 f32

    for (int ks = 0; ks < KSTEPS; ++ks) {
        const int kh = ks / 12;
        const int t2 = ks - kh * 12;
        const int kw = t2 >> 2;
        const int c0 = (t2 & 3) * 32;
        const long aIdx = aBaseIdx + (kh * 64 + kw) * CIN + c0 + cOff;
        const long bIdx = bBaseIdx + ks * 32;

        // ---- global fetch (before barrier so loads overlap the wait) ----
        float4 a0, a1, a2, a3, b0, b1, b2, b3, b4, b5, b6, b7;
        v8h ah0, ah1, al0, al1, bh0, bh1, bh2, bh3, bl0, bl1, bl2, bl3;
        if constexpr (MODE == 0) {
            const float* aPtr = in + aIdx;
            const float* bPtr = w + bIdx;
            a0 = ((const float4*)aPtr)[0]; a1 = ((const float4*)aPtr)[1];
            a2 = ((const float4*)aPtr)[2]; a3 = ((const float4*)aPtr)[3];
            b0 = ((const float4*)bPtr)[0]; b1 = ((const float4*)bPtr)[1];
            b2 = ((const float4*)bPtr)[2]; b3 = ((const float4*)bPtr)[3];
            b4 = ((const float4*)bPtr)[4]; b5 = ((const float4*)bPtr)[5];
            b6 = ((const float4*)bPtr)[6]; b7 = ((const float4*)bPtr)[7];
        } else {
            ah0 = ((const v8h*)(inHi + aIdx))[0]; ah1 = ((const v8h*)(inHi + aIdx))[1];
            al0 = ((const v8h*)(inLo + aIdx))[0]; al1 = ((const v8h*)(inLo + aIdx))[1];
            bh0 = ((const v8h*)(wHi + bIdx))[0];  bh1 = ((const v8h*)(wHi + bIdx))[1];
            bh2 = ((const v8h*)(wHi + bIdx))[2];  bh3 = ((const v8h*)(wHi + bIdx))[3];
            bl0 = ((const v8h*)(wLo + bIdx))[0];  bl1 = ((const v8h*)(wLo + bIdx))[1];
            bl2 = ((const v8h*)(wLo + bIdx))[2];  bl3 = ((const v8h*)(wLo + bIdx))[3];
        }

        // Prefetch next k-step (global_prefetch_b8).
        if (ks + 1 < KSTEPS) {
            const int ks2 = ks + 1;
            const int kh2 = ks2 / 12;
            const int u2  = ks2 - kh2 * 12;
            const long aN = aBaseIdx + (kh2 * 64 + (u2 >> 2)) * CIN + (u2 & 3) * 32 + cOff;
            const long bN = bBaseIdx + ks2 * 32;
            if constexpr (MODE == 0) {
                __builtin_prefetch(in + aN, 0, 3);
                __builtin_prefetch(w + bN, 0, 3);
            } else {
                __builtin_prefetch(inHi + aN, 0, 3);
                __builtin_prefetch(inLo + aN, 0, 3);
                __builtin_prefetch(wHi + bN, 0, 3);
                __builtin_prefetch(wLo + bN, 0, 3);
            }
        }

        __syncthreads();   // previous iteration's LDS reads complete

        if constexpr (MODE == 0) {
            float av[16] = { a0.x, a0.y, a0.z, a0.w,  a1.x, a1.y, a1.z, a1.w,
                             a2.x, a2.y, a2.z, a2.w,  a3.x, a3.y, a3.z, a3.w };
            float bv0[16] = { b0.x, b0.y, b0.z, b0.w,  b1.x, b1.y, b1.z, b1.w,
                              b2.x, b2.y, b2.z, b2.w,  b3.x, b3.y, b3.z, b3.w };
            float bv1[16] = { b4.x, b4.y, b4.z, b4.w,  b5.x, b5.y, b5.z, b5.w,
                              b6.x, b6.y, b6.z, b6.w,  b7.x, b7.y, b7.z, b7.w };
            split_store16(av,  &AsHi[aRow * LDSS + cOff], &AsLo[aRow * LDSS + cOff]);
            split_store16(bv0, &BsHi[t * LDSS],           &BsLo[t * LDSS]);
            split_store16(bv1, &BsHi[t * LDSS + 16],      &BsLo[t * LDSS + 16]);
        } else {
            _Float16* ad = &AsHi[aRow * LDSS + cOff];
            _Float16* al = &AsLo[aRow * LDSS + cOff];
            ((v8h*)ad)[0] = ah0; ((v8h*)ad)[1] = ah1;
            ((v8h*)al)[0] = al0; ((v8h*)al)[1] = al1;
            _Float16* bd = &BsHi[t * LDSS];
            _Float16* bl = &BsLo[t * LDSS];
            ((v8h*)bd)[0] = bh0; ((v8h*)bd)[1] = bh1;
            ((v8h*)bd)[2] = bh2; ((v8h*)bd)[3] = bh3;
            ((v8h*)bl)[0] = bl0; ((v8h*)bl)[1] = bl1;
            ((v8h*)bl)[2] = bl2; ((v8h*)bl)[3] = bl3;
        }

        __syncthreads();   // tiles visible to all waves

        // ---- A fragments up front (reused across all 4 B-subtiles) ----
        // A 16x32 f16: lane group g holds K = [16g..16g+15].
        v16h Ah[4], Al[4];
#pragma unroll
        for (int i = 0; i < 4; ++i) {
            const int m = wm * 64 + i * 16 + mlane;
            const _Float16* ph = &AsHi[m * LDSS];
            const _Float16* pl = &AsLo[m * LDSS];
            Ah[i] = cat8(*(const v8h*)(ph + 16 * g), *(const v8h*)(ph + 16 * g + 8));
            Al[i] = cat8(*(const v8h*)(pl + 16 * g), *(const v8h*)(pl + 16 * g + 8));
        }
        // ---- B fragments per j to bound live registers ----
        // B (K-major per column): lane group g holds K = [16g..16g+15].
#pragma unroll
        for (int j = 0; j < 4; ++j) {
            const int f = wn * 64 + j * 16 + mlane;
            const _Float16* ph = &BsHi[f * LDSS];
            const _Float16* pl = &BsLo[f * LDSS];
            v16h Bh = cat8(*(const v8h*)(ph + 16 * g), *(const v8h*)(ph + 16 * g + 8));
            v16h Bl = cat8(*(const v8h*)(pl + 16 * g), *(const v8h*)(pl + 16 * g + 8));
#pragma unroll
            for (int i = 0; i < 4; ++i) {
                acc[i][j] = __builtin_amdgcn_wmma_f32_16x16x32_f16(
                    false, Ah[i], false, Bh, (short)0, acc[i][j], false, false);
                acc[i][j] = __builtin_amdgcn_wmma_f32_16x16x32_f16(
                    false, Ah[i], false, Bl, (short)0, acc[i][j], false, false);
                acc[i][j] = __builtin_amdgcn_wmma_f32_16x16x32_f16(
                    false, Al[i], false, Bh, (short)0, acc[i][j], false, false);
            }
        }
    }

    // ---- epilogue: C layout (VGPR e -> row e + 8g; lanes 0-15 = cols), add bias ----
#pragma unroll
    for (int j = 0; j < 4; ++j) {
        const int fc = wn * 64 + j * 16 + mlane;
        const float bv = bias[fc];
#pragma unroll
        for (int i = 0; i < 4; ++i) {
            const int rbase = m0 + wm * 64 + i * 16 + g * 8;
#pragma unroll
            for (int e = 0; e < 8; ++e) {
                out[(long)(rbase + e) * FOUT + fc] = acc[i][j][e] + bv;
            }
        }
    }
}

extern "C" void kernel_launch(void* const* d_in, const int* in_sizes, int n_in,
                              void* d_out, int out_size, void* d_ws, size_t ws_size,
                              hipStream_t stream) {
    (void)in_sizes; (void)n_in; (void)out_size;
    const float* in   = (const float*)d_in[0];   // (32,64,64,128) fp32
    const float* w    = (const float*)d_in[1];   // (256,1152) fp32
    const float* bias = (const float*)d_in[2];   // (256,) fp32
    float* out = (float*)d_out;                  // (32,62,62,256) fp32

    // ws layout (MODE 1): [inHi | inLo | wHi | wLo], all f16.
    const size_t needed = ((size_t)NIN + (size_t)NW) * 2 * sizeof(_Float16);
    _Float16* inHi = (_Float16*)d_ws;
    _Float16* inLo = inHi + NIN;
    _Float16* wHi  = inLo + NIN;
    _Float16* wLo  = wHi + NW;

    dim3 grid(961, 1, 1);
    dim3 block(256, 1, 1);

    if (ws_size >= needed) {
        // Pre-split every operand exactly once, then run the pure-f16 GEMM.
        split_f32_to_f16x2<<<NIN / 8 / 256, 256, 0, stream>>>(in, inHi, inLo, NIN / 8);
        split_f32_to_f16x2<<<NW / 8 / 256, 256, 0, stream>>>(w, wHi, wLo, NW / 8);
        hipLaunchKernelGGL((conv3x3_wmma_f16split_kernel<1>), grid, block, 0, stream,
                           in, w, bias, out, inHi, inLo, wHi, wLo);
    } else {
        hipLaunchKernelGGL((conv3x3_wmma_f16split_kernel<0>), grid, block, 0, stream,
                           in, w, bias, out, inHi, inLo, wHi, wLo);
    }
}
